// Conv2dSDK_QR_79912161509723
// MI455X (gfx1250) — matrix operand order repaired and checked by reading, not compiled, not run on hardware
//
#include <hip/hip_runtime.h>

typedef __bf16 bf16_t;
typedef __attribute__((ext_vector_type(16))) __bf16 v16bf;
typedef __attribute__((ext_vector_type(8)))  __bf16 v8bf;
typedef __attribute__((ext_vector_type(8)))  float   v8f;
typedef __attribute__((ext_vector_type(4)))  float   v4f;
typedef __attribute__((ext_vector_type(4)))  unsigned int v4u;
typedef __attribute__((ext_vector_type(8)))  int     v8i;
typedef __attribute__((ext_vector_type(4)))  int     v4i;

namespace {
constexpr int kN  = 32;     // batch
constexpr int kC  = 128;    // in channels
constexpr int kH  = 64;
constexpr int kW  = 64;
constexpr int kL  = 1024;   // rows per image (32*32)
constexpr int kF  = 2048;   // C * PW * PW
constexpr int kR  = 256;    // rank
constexpr int kM2 = 512;    // (s*s)*OC
constexpr int kOC = 128;

constexpr int MT = 64;      // l-rows per workgroup
constexpr int KB = 64;      // f K-block for GEMM1
constexpr int MC = 64;      // m-chunk for GEMM2

// LDS layout in bf16 elements (dynamic segment base == LDS offset 0)
constexpr int T_OFF = 0;                 // T_lds[MT][kR]
constexpr int T_SZ  = MT * kR;           // 16384
constexpr int A_OFF = T_OFF + T_SZ;      // A_lds[MT][KB]   (phase 1)
constexpr int A_SZ  = MT * KB;           // 4096
constexpr int B_OFF = A_OFF + A_SZ;      // Bs[kR][KB]      (R^T, column-major B)
constexpr int B_SZ  = kR * KB;           // 16384
constexpr int Q_OFF = A_OFF;             // Qb[MC][kR]      (phase 2, aliases A+B)
constexpr int LDS_ELEMS = T_SZ + A_SZ + B_SZ;   // 36864 elems = 73728 bytes

constexpr size_t RB_ELEMS = (size_t)kR * kF;     // bf16 copy of R in ws
constexpr size_t QB_ELEMS = (size_t)kM2 * kR;    // bf16 copy of Q in ws
constexpr size_t WS_NEED  = (RB_ELEMS + QB_ELEMS) * sizeof(bf16_t);  // ~1.25 MiB
} // namespace

__device__ inline v8f vzero8() {
  v8f z;
#pragma unroll
  for (int i = 0; i < 8; ++i) z[i] = 0.0f;
  return z;
}

// A fragment (16x32 bf16): lanes 0-15 hold K = k0+{0..15}, lanes 16-31 hold K = k0+{16..31}.
__device__ inline v16bf frag_a16(const bf16_t* p, int hi) {
  union { v16bf v; v8bf h[2]; } u;
  const v8bf* q = (const v8bf*)(p + hi * 16);
  u.h[0] = q[0];
  u.h[1] = q[1];
  return u.v;
}

// B fragment (32x16 bf16), column-major storage: lane holds 16 contiguous K for its column.
__device__ inline v16bf frag_b16(const bf16_t* colbase, int hi) {
  union { v16bf v; v8bf h[2]; } u;
  const v8bf* q = (const v8bf*)(colbase + hi * 16);
  u.h[0] = q[0];
  u.h[1] = q[1];
  return u.v;
}

// Issue a 2-D TDM tile load: tensor[tensor_d1][tensor_d0] bf16 (row stride stride0),
// tile [tile_d1][tile_d0] starting at gaddr, packed contiguously into LDS at lds_byte.
// D# layout per CDNA5 ISA 8.3/8.4: group0 = {flags, lds_addr, global_addr, type},
// group1 = {data_size, tensor dims, tile dims, dim0 stride}; groups 2/3 zero (2-D).
// Toolchain here is the clang-23 6-arg builtin: (v4u, v8i, v4i, v4i, v8i, imm cpol).
__device__ inline void tdm_load_2d(unsigned lds_byte, const bf16_t* gaddr,
                                   unsigned tensor_d0, unsigned tensor_d1,
                                   unsigned stride0, unsigned tile_d0,
                                   unsigned tile_d1) {
  unsigned long long ga = (unsigned long long)(size_t)gaddr;
  v4u g0;
  g0[0] = 1u;                                              // count=1, user D#
  g0[1] = lds_byte;                                        // lds_addr
  g0[2] = (unsigned)ga;                                    // global_addr[31:0]
  g0[3] = ((unsigned)(ga >> 32) & 0x01FFFFFFu) | (2u << 30); // addr[56:32] | type=2
  v8i g1;
  g1[0] = (int)(1u << 16);                                 // data_size=1 (2B), mask=0
  g1[1] = (int)((tensor_d0 & 0xFFFFu) << 16);              // tensor_dim0[15:0] @ bit48
  g1[2] = (int)(((tensor_d0 >> 16) & 0xFFFFu) | ((tensor_d1 & 0xFFFFu) << 16));
  g1[3] = (int)(((tensor_d1 >> 16) & 0xFFFFu) | ((tile_d0 & 0xFFFFu) << 16));
  g1[4] = (int)(tile_d1 & 0xFFFFu);                        // tile_dim1, tile_dim2=0
  g1[5] = (int)stride0;                                    // tensor_dim0_stride[31:0]
  g1[6] = 0;                                               // stride0 hi / stride1 lo
  g1[7] = 0;
  v4i gz;
  gz[0] = 0; gz[1] = 0; gz[2] = 0; gz[3] = 0;
  v8i gz8;
#pragma unroll
  for (int i = 0; i < 8; ++i) gz8[i] = 0;
  __builtin_amdgcn_tensor_load_to_lds(g0, g1, gz, gz, gz8, 0);
}

// One-shot f32 -> bf16 conversion of R then Q into the workspace.
__global__ void cvt_rq_kernel(const float* __restrict__ Rg,
                              const float* __restrict__ Qg,
                              bf16_t* __restrict__ ws) {
  size_t i = ((size_t)blockIdx.x * blockDim.x + threadIdx.x) * 4;
  const float* src = (i < RB_ELEMS) ? (Rg + i) : (Qg + (i - RB_ELEMS));
  v4f v = *(const v4f*)src;
  bf16_t* d = ws + i;
  d[0] = (bf16_t)v.x; d[1] = (bf16_t)v.y; d[2] = (bf16_t)v.z; d[3] = (bf16_t)v.w;
}

template <bool USE_TDM>
__global__ __launch_bounds__(256, 1)
void sdk_conv_lowrank_kernel(const float* __restrict__ xg,
                             const float* __restrict__ Qg,
                             const float* __restrict__ Rg,
                             const bf16_t* __restrict__ wsb,  // bf16 R then Q (TDM path)
                             float* __restrict__ outg) {
  extern __shared__ bf16_t lds[];
  bf16_t* Tl = lds + T_OFF;
  bf16_t* Al = lds + A_OFF;
  bf16_t* Bl = lds + B_OFF;
  bf16_t* Ql = lds + Q_OFF;

  const int tid  = threadIdx.x;
  const int lane = tid & 31;
  const int wave = tid >> 5;
  const int hi   = lane >> 4;   // which K-half this lane holds
  const int ln   = lane & 15;   // row (A) / column (B,D) within 16

  const int blk   = blockIdx.x;     // 0..511
  const int n     = blk >> 4;       // image
  const int ltile = blk & 15;
  const int lbase = ltile * MT;

  const int wm = wave & 3;          // 16-row M block within the 64-row tile
  const int wn = wave >> 2;         // phase-1: which 128-col half of rank dim

  // ---------------- Phase 1: T[l, r] = sum_f im2col(x)[l, f] * R[r, f] ----------------
  v8f acc[8];
#pragma unroll
  for (int i = 0; i < 8; ++i) acc[i] = vzero8();

  // per-thread im2col assignment: 4 threads per row, each covers one channel (16 f's)
  const int arow = tid >> 2;        // 0..63
  const int agrp = tid & 3;
  const int al   = lbase + arow;
  const int aa   = al >> 5;         // a = l / 32
  const int ab   = al & 31;         // b = l % 32
  const int y0   = 2 * aa - 1;      // pad of 1
  const int x0   = 2 * ab - 1;

  for (int kb = 0; kb < kF / KB; ++kb) {
    const int fbase = kb * KB;

    // B tile first: DMA overlaps the im2col VALU gather below.
    if constexpr (USE_TDM) {
      if (wave == 0) {
        // tile [kR rows][KB cols] of bf16-R [kR][kF], packed -> Bl[r][kk]
        tdm_load_2d((unsigned)(B_OFF * sizeof(bf16_t)), wsb + fbase,
                    kF, kR, kF, KB, kR);
      }
    } else {
      const float* rp = Rg + (size_t)tid * kF + fbase;
      bf16_t* dst = Bl + tid * KB;
#pragma unroll
      for (int kk = 0; kk < KB; kk += 4) {
        v4f rv = *(const v4f*)(rp + kk);
        dst[kk + 0] = (bf16_t)rv.x;
        dst[kk + 1] = (bf16_t)rv.y;
        dst[kk + 2] = (bf16_t)rv.z;
        dst[kk + 3] = (bf16_t)rv.w;
      }
    }

    // A tile: im2col gather, f32 -> bf16 (x is L2-resident; zero-fill the halo)
    {
      const int c = (fbase >> 4) + agrp;
      const float* xc = xg + ((size_t)(n * kC + c) * kH) * kW;
      bf16_t* dst = Al + arow * KB + agrp * 16;
#pragma unroll
      for (int e = 0; e < 16; ++e) {
        const int iy = e >> 2, jx = e & 3;
        const int y = y0 + iy, x = x0 + jx;
        float v = 0.0f;
        if ((unsigned)y < (unsigned)kH && (unsigned)x < (unsigned)kW) v = xc[y * kW + x];
        dst[e] = (bf16_t)v;
      }
    }

    if constexpr (USE_TDM) {
      if (wave == 0) __builtin_amdgcn_s_wait_tensorcnt(0);
    }
    __syncthreads();

#pragma unroll
    for (int ks = 0; ks < KB; ks += 32) {
      const v16bf afrag = frag_a16(Al + (wm * 16 + ln) * KB + ks, hi);
#pragma unroll
      for (int nt = 0; nt < 8; ++nt) {
        const int col = wn * 128 + nt * 16 + ln;
        const v16bf bfrag = frag_b16(Bl + col * KB + ks, hi);
        acc[nt] = __builtin_amdgcn_wmma_f32_16x16x32_bf16(
            false, afrag, false, bfrag, (short)0, acc[nt], false, false);
      }
    }
    __syncthreads();
  }

  // park T in LDS as bf16 (never touches HBM)
#pragma unroll
  for (int nt = 0; nt < 8; ++nt) {
    const int col = wn * 128 + nt * 16 + ln;
#pragma unroll
    for (int v = 0; v < 8; ++v) {
      const int ml = v + hi * 8;   // D layout: lanes 0-15 -> M=v, lanes 16-31 -> M=v+8
      Tl[(wm * 16 + ml) * kR + col] = (bf16_t)acc[nt][v];
    }
  }
  __syncthreads();

  // ---------------- Phase 2: out[l, m] = sum_r T[l, r] * Q[m, r] ----------------
  const int qrow = tid >> 2;   // 0..63 rows of the Q chunk
  const int qq   = tid & 3;    // quarter of the 256 r's
  const int wn2  = wave >> 2;  // which 32-col half of the 64-col chunk

  for (int mc = 0; mc < kM2 / MC; ++mc) {
    const int mbase = mc * MC;
    if constexpr (USE_TDM) {
      if (wave == 0) {
        // tile [MC rows][kR cols] of bf16-Q [kM2][kR], packed -> Ql[cc][r]
        tdm_load_2d((unsigned)(Q_OFF * sizeof(bf16_t)),
                    wsb + RB_ELEMS + (size_t)mbase * kR,
                    kR, kM2, kR, kR, MC);
        __builtin_amdgcn_s_wait_tensorcnt(0);
      }
    } else {
      const float* qp = Qg + (size_t)(mbase + qrow) * kR + qq * 64;
      bf16_t* dst = Ql + qrow * kR + qq * 64;
#pragma unroll
      for (int kk = 0; kk < 64; kk += 4) {
        v4f qv = *(const v4f*)(qp + kk);
        dst[kk + 0] = (bf16_t)qv.x;
        dst[kk + 1] = (bf16_t)qv.y;
        dst[kk + 2] = (bf16_t)qv.z;
        dst[kk + 3] = (bf16_t)qv.w;
      }
    }
    __syncthreads();

    v8f acc2[2];
    acc2[0] = vzero8();
    acc2[1] = vzero8();
#pragma unroll
    for (int k0 = 0; k0 < kR; k0 += 32) {
      const v16bf afrag = frag_a16(Tl + (wm * 16 + ln) * kR + k0, hi);
#pragma unroll
      for (int t = 0; t < 2; ++t) {
        const int colc = wn2 * 32 + t * 16 + ln;
        const v16bf bfrag = frag_b16(Ql + colc * kR + k0, hi);
        acc2[t] = __builtin_amdgcn_wmma_f32_16x16x32_bf16(
            false, afrag, false, bfrag, (short)0, acc2[t], false, false);
      }
    }

    // fused pixel-shuffle scatter: m = (p*2+q)*128 + oc ; h = 2a+p ; w = 2b+q
#pragma unroll
    for (int t = 0; t < 2; ++t) {
      const int m  = mbase + wn2 * 32 + t * 16 + ln;
      const int p  = m >> 8;
      const int q  = (m >> 7) & 1;
      const int oc = m & 127;
#pragma unroll
      for (int v = 0; v < 8; ++v) {
        const int ml = v + hi * 8;
        const int l  = lbase + wm * 16 + ml;
        const int a  = l >> 5, b = l & 31;
        const int h  = 2 * a + p, w = 2 * b + q;
        outg[(((size_t)n * kOC + oc) * kH + h) * kW + w] = acc2[t][v];
      }
    }
    __syncthreads();
  }
}

extern "C" void kernel_launch(void* const* d_in, const int* in_sizes, int n_in,
                              void* d_out, int out_size, void* d_ws, size_t ws_size,
                              hipStream_t stream) {
  const float* x = (const float*)d_in[0];
  const float* Q = (const float*)d_in[1];
  const float* R = (const float*)d_in[2];
  float* out = (float*)d_out;
  (void)in_sizes; (void)n_in; (void)out_size;

  dim3 grid(kN * (kL / MT));   // 512 workgroups
  dim3 block(256);             // 8 wave32s
  size_t shmem = (size_t)LDS_ELEMS * sizeof(bf16_t);   // 73728 B of the 320 KB/WGP

  if (ws_size >= WS_NEED) {
    // one-shot bf16 conversion of the weight factors, then TDM-staged main kernel
    bf16_t* wsb = (bf16_t*)d_ws;
    const size_t total4 = (RB_ELEMS + QB_ELEMS) / 4;       // 163840 threads
    hipLaunchKernelGGL(cvt_rq_kernel, dim3((unsigned)(total4 / 256)), dim3(256),
                       0, stream, R, Q, wsb);
    hipLaunchKernelGGL((sdk_conv_lowrank_kernel<true>), grid, block, shmem, stream,
                       x, Q, R, (const bf16_t*)wsb, out);
  } else {
    hipLaunchKernelGGL((sdk_conv_lowrank_kernel<false>), grid, block, shmem, stream,
                       x, Q, R, (const bf16_t*)nullptr, out);
  }
}
